// QuantLinear_65781719106079
// MI455X (gfx1250) — 8-bit matrix operand order repaired and checked by reading, compiled, not run on hardware
//
#include <hip/hip_runtime.h>

typedef __attribute__((ext_vector_type(4))) int   v4i;
typedef __attribute__((ext_vector_type(8))) int   v8i;
typedef __attribute__((ext_vector_type(4))) float v4f;

#define QMAXI 127
#define QMAXF 127.0f


template <bool PACK_B>
__global__ void __launch_bounds__(256)
quant_pack_kernel(const float* __restrict__ src,
                  char* __restrict__ dst,
                  float* __restrict__ scales,
                  int cols)
{
    __shared__ float red[256];
    const int row = blockIdx.x;
    const int t   = threadIdx.x;
    const int KT  = cols >> 6;

    const float* p = src + (size_t)row * cols;

    float m = 0.0f;
    for (int i = t * 4; i < cols; i += 256 * 4) {
        v4f v = *(const v4f*)(p + i);
        m = fmaxf(m, fabsf(v.x));
        m = fmaxf(m, fabsf(v.y));
        m = fmaxf(m, fabsf(v.z));
        m = fmaxf(m, fabsf(v.w));
    }
    red[t] = m;
    __syncthreads();
    for (int s = 128; s > 0; s >>= 1) {
        if (t < s) red[t] = fmaxf(red[t], red[t + s]);
        __syncthreads();
    }
    const float scale = fmaxf(red[0] / QMAXF, 1e-8f);
    const float inv   = 1.0f / scale;
    if (t == 0) scales[row] = scale;

    const int rt = row >> 4;
    const int rr = row & 15;

    for (int i = t * 4; i < cols; i += 256 * 4) {
        v4f v = *(const v4f*)(p + i);
        int q0 = min(max(__float2int_rn(v.x * inv), -QMAXI), QMAXI);
        int q1 = min(max(__float2int_rn(v.y * inv), -QMAXI), QMAXI);
        int q2 = min(max(__float2int_rn(v.z * inv), -QMAXI), QMAXI);
        int q3 = min(max(__float2int_rn(v.w * inv), -QMAXI), QMAXI);
        int packed = (q0 & 0xff) | ((q1 & 0xff) << 8) |
                     ((q2 & 0xff) << 16) | ((q3 & 0xff) << 24);

        const int kt   = i >> 6;
        const int koff = i & 63;
        int half, vreg;
        if (PACK_B) {
            const int g   = koff >> 5;
            const int rem = koff & 31;
            half = rem >> 4;
            vreg = g * 4 + ((rem & 15) >> 2);
        } else {
            const int q    = koff >> 5;
            const int rem2 = koff & 31;
            half = rem2 >> 4;
            vreg = q * 4 + ((rem2 & 15) >> 2);
        }
        const int lane = half * 16 + rr;
        const size_t off = ((size_t)(rt * KT + kt) * 32 + lane) * 32 + vreg * 4;
        *(int*)(dst + off) = packed;
    }
}

__device__ __forceinline__ v8i load_frag(const char* base) {
    v4i lo = *(const v4i*)(base);
    v4i hi = *(const v4i*)(base + 16);
    v8i f;
    f[0] = lo[0]; f[1] = lo[1]; f[2] = lo[2]; f[3] = lo[3];
    f[4] = hi[0]; f[5] = hi[1]; f[6] = hi[2]; f[7] = hi[3];
    return f;
}

__global__ void __launch_bounds__(256)
gemm_i8_wmma_kernel(const char* __restrict__ xq,
                    const float* __restrict__ sx,
                    const char* __restrict__ wq,
                    const float* __restrict__ sw,
                    const float* __restrict__ bias,
                    float* __restrict__ out,
                    int M, int N, int K)
{
    const int lane = threadIdx.x & 31;
    const int wave = threadIdx.x >> 5;
    const int wm   = wave & 3;
    const int wn   = wave >> 2;
    const int half = lane >> 4;
    const int l16  = lane & 15;
    const int KT   = K >> 6;

    const int m0 = blockIdx.x * 128 + wm * 32;
    const int n0 = blockIdx.y * 128 + wn * 64;

    const char* aBase0 = xq + ((size_t)((m0 >> 4) + 0) * KT) * 1024 + lane * 32;
    const char* aBase1 = xq + ((size_t)((m0 >> 4) + 1) * KT) * 1024 + lane * 32;
    const char* bBase0 = wq + ((size_t)((n0 >> 4) + 0) * KT) * 1024 + lane * 32;
    const char* bBase1 = wq + ((size_t)((n0 >> 4) + 1) * KT) * 1024 + lane * 32;
    const char* bBase2 = wq + ((size_t)((n0 >> 4) + 2) * KT) * 1024 + lane * 32;
    const char* bBase3 = wq + ((size_t)((n0 >> 4) + 3) * KT) * 1024 + lane * 32;

    v8i acc00 = v8i{0}, acc01 = v8i{0}, acc02 = v8i{0}, acc03 = v8i{0};
    v8i acc10 = v8i{0}, acc11 = v8i{0}, acc12 = v8i{0}, acc13 = v8i{0};

    for (int kt = 0; kt < KT; ++kt) {
        const size_t ko = (size_t)kt * 1024;
        v8i a0 = load_frag(aBase0 + ko);
        v8i a1 = load_frag(aBase1 + ko);
        v8i b0 = load_frag(bBase0 + ko);
        v8i b1 = load_frag(bBase1 + ko);
        v8i b2 = load_frag(bBase2 + ko);
        v8i b3 = load_frag(bBase3 + ko);

        acc00 = __builtin_amdgcn_wmma_i32_16x16x64_iu8(true, a0, true, b0, acc00, false, false);
        acc10 = __builtin_amdgcn_wmma_i32_16x16x64_iu8(true, a1, true, b0, acc10, false, false);
        acc01 = __builtin_amdgcn_wmma_i32_16x16x64_iu8(true, a0, true, b1, acc01, false, false);
        acc11 = __builtin_amdgcn_wmma_i32_16x16x64_iu8(true, a1, true, b1, acc11, false, false);
        acc02 = __builtin_amdgcn_wmma_i32_16x16x64_iu8(true, a0, true, b2, acc02, false, false);
        acc12 = __builtin_amdgcn_wmma_i32_16x16x64_iu8(true, a1, true, b2, acc12, false, false);
        acc03 = __builtin_amdgcn_wmma_i32_16x16x64_iu8(true, a0, true, b3, acc03, false, false);
        acc13 = __builtin_amdgcn_wmma_i32_16x16x64_iu8(true, a1, true, b3, acc13, false, false);
    }

    float sxv0[8], sxv1[8];
#pragma unroll
    for (int r = 0; r < 8; ++r) {
        sxv0[r] = sx[m0 +      r + half * 8];
        sxv1[r] = sx[m0 + 16 + r + half * 8];
    }

    v8i* accs[2][4] = {{&acc00, &acc01, &acc02, &acc03},
                       {&acc10, &acc11, &acc12, &acc13}};
#pragma unroll
    for (int j = 0; j < 4; ++j) {
        const int col  = n0 + j * 16 + l16;
        const float swn = sw[col];
        const float bn  = bias[col];
#pragma unroll
        for (int r = 0; r < 8; ++r) {
            const int row0 = m0 +      r + half * 8;
            const int row1 = m0 + 16 + r + half * 8;
            out[(size_t)row0 * N + col] = (float)((*accs[0][j])[r]) * sxv0[r] * swn + bn;
            out[(size_t)row1 * N + col] = (float)((*accs[1][j])[r]) * sxv1[r] * swn + bn;
        }
    }
}

extern "C" void kernel_launch(void* const* d_in, const int* in_sizes, int n_in,
                              void* d_out, int out_size, void* d_ws, size_t ws_size,
                              hipStream_t stream) {
    const float* x    = (const float*)d_in[0];
    const float* w    = (const float*)d_in[1];
    const float* bias = (const float*)d_in[2];
    float* out = (float*)d_out;

    const int N = in_sizes[2];
    const int K = in_sizes[1] / N;
    const int M = (int)((long long)in_sizes[0] / K);

    char*  ws  = (char*)d_ws;
    char*  xq  = ws;
    char*  wqp = xq + (size_t)M * K;
    float* sx  = (float*)(wqp + (size_t)N * K);
    float* sw  = sx + M;

    quant_pack_kernel<false><<<M, 256, 0, stream>>>(x, xq, sx, K);
    quant_pack_kernel<true ><<<N, 256, 0, stream>>>(w, wqp, sw, K);

    dim3 grid(M / 128, N / 128);
    gemm_i8_wmma_kernel<<<grid, 256, 0, stream>>>(xq, sx, wqp, sw, bias, out,
                                                  M, N, K);
}
